// RNNProgressNet_16947940950590
// MI455X (gfx1250) — matrix operand order repaired and checked by reading, not compiled, not run on hardware
//
#include <hip/hip_runtime.h>

typedef float  v2f  __attribute__((ext_vector_type(2)));
typedef float  v8f  __attribute__((ext_vector_type(8)));
typedef __bf16 v16bf __attribute__((ext_vector_type(16)));
typedef __bf16 v8bf  __attribute__((ext_vector_type(8)));

union BF16x16 { v16bf v; v8bf h[2]; };

__device__ __forceinline__ v8f wmma4(v2f a, v2f b, v8f c) {
  // v_wmma_f32_16x16x4_f32
  return __builtin_amdgcn_wmma_f32_16x16x4_f32(false, a, false, b, (short)0, c, false, false);
}
__device__ __forceinline__ v8f wmma_bf(v16bf a, v16bf b, v8f c) {
  // v_wmma_f32_16x16x32_bf16 (K=32 per instruction, f32 accumulate)
  return __builtin_amdgcn_wmma_f32_16x16x32_bf16(false, a, false, b, (short)0, c, false, false);
}

__device__ __forceinline__ float sigm(float x) { return 1.0f / (1.0f + __expf(-x)); }

// ---------------- workspace layout (float offsets) ----------------
// W1B: bf16[96 x 4096] packed fragment-native: idx = kb*131072 + n*32 + half*16 + kk
// W7B: bf16[4096 x 64] packed fragment-native: idx = kb*2048  + n*32 + half*16 + kk
#define WS_W1B     0        // 393216 bf16 = 196608 floats
#define WS_W7B     196608   // 262144 bf16 = 131072 floats
#define WS_APAD    327680   // 2048 x 96 f32 (pooled, K padded)
#define WS_E       524288   // 2048 x 64 f32
#define WS_X1      655360   // 2048 x 256 f32
#define WS_WIH1T   1179648  // 64 x 256 f32
#define WS_WHH1T   1196032  // 64 x 256 f32
#define WS_WIH2T   1212416  // 64 x 128 f32
#define WS_WHH2T   1220608  // 32 x 128 f32
#define WS_FWT     1224704  // 32 x 90 f32
#define WS_B1C     1227584  // 256 f32
#define WS_B2C     1227840  // 128 f32

// ---------------- K0: weight pack / transpose / bias-combine ----------------
__global__ void prep_kernel(const float* __restrict__ W1, const float* __restrict__ W7,
                            const float* __restrict__ wih1, const float* __restrict__ whh1,
                            const float* __restrict__ wih2, const float* __restrict__ whh2,
                            const float* __restrict__ fw,
                            const float* __restrict__ bih1, const float* __restrict__ bhh1,
                            const float* __restrict__ bih2, const float* __restrict__ bhh2,
                            float* __restrict__ ws) {
  int idx = blockIdx.x * 256 + threadIdx.x;
  __bf16* W1B = (__bf16*)(ws + WS_W1B);
  __bf16* W7B = (__bf16*)(ws + WS_W7B);
  if (idx < 393216) { // W1B fragment-native pack of spp_fc_w^T, K padded 90->96
    int kb = idx >> 17, rem = idx & 131071;
    int n = rem >> 5, t = rem & 31, half = t >> 4, kk = t & 15;
    int k = kb * 32 + ((kk & 8) << 1) + half * 8 + (kk & 7);
    W1B[idx] = (k < 90) ? (__bf16)W1[(size_t)n * 90 + k] : (__bf16)0.0f;
    return;
  }
  idx -= 393216;
  if (idx < 262144) { // W7B fragment-native pack of fc7_w^T
    int kb = idx >> 11, rem = idx & 2047;
    int n = rem >> 5, t = rem & 31, half = t >> 4, kk = t & 15;
    int k = kb * 32 + ((kk & 8) << 1) + half * 8 + (kk & 7);
    W7B[idx] = (__bf16)W7[(size_t)n * 4096 + k];
    return;
  }
  idx -= 262144;
  if (idx < 16384) { int k = idx >> 8, j = idx & 255; ws[WS_WIH1T + idx] = wih1[j * 64 + k]; return; }
  idx -= 16384;
  if (idx < 16384) { int k = idx >> 8, j = idx & 255; ws[WS_WHH1T + idx] = whh1[j * 64 + k]; return; }
  idx -= 16384;
  if (idx < 8192)  { int k = idx >> 7, j = idx & 127; ws[WS_WIH2T + idx] = wih2[j * 64 + k]; return; }
  idx -= 8192;
  if (idx < 4096)  { int k = idx >> 7, j = idx & 127; ws[WS_WHH2T + idx] = whh2[j * 32 + k]; return; }
  idx -= 4096;
  if (idx < 2880)  { int k = idx / 90, j = idx % 90; ws[WS_FWT + idx] = fw[j * 32 + k]; return; }
  idx -= 2880;
  if (idx < 256)   { ws[WS_B1C + idx] = bih1[idx] + bhh1[idx]; return; }
  idx -= 256;
  if (idx < 128)   { ws[WS_B2C + idx] = bih2[idx] + bhh2[idx]; }
}

// ---------------- K1: SPP pooling (HBM-bound, read frames exactly once) ----
__global__ __launch_bounds__(256) void spp_kernel(const float* __restrict__ frames,
                                                  float* __restrict__ dout,
                                                  float* __restrict__ Apad) {
  __shared__ float g8[432]; // 3 channels x 12x12 grid of 8x8 maxima
  const int img = blockIdx.x, tid = threadIdx.x;
  const float* F = frames + (size_t)img * 27648;
  for (int c = tid; c < 432; c += 256) {
    int ch = c / 144, cell = c % 144, gy = cell / 12, gx = cell % 12;
    const float* p = F + ch * 9216 + gy * 768 + gx * 8;
    float m = -__builtin_inff();
    #pragma unroll
    for (int r = 0; r < 8; ++r) {
      float4 a = *(const float4*)(p + r * 96);
      float4 b = *(const float4*)(p + r * 96 + 4);
      m = fmaxf(m, fmaxf(fmaxf(fmaxf(a.x, a.y), fmaxf(a.z, a.w)),
                         fmaxf(fmaxf(b.x, b.y), fmaxf(b.z, b.w))));
    }
    g8[c] = m;
  }
  __syncthreads();
  if (tid < 90) {
    int f = tid, ch, y0, x0, ny;
    if (f < 48)      { ch = f / 16; int r = f % 16; y0 = (r / 4) * 3; x0 = (r % 4) * 3; ny = 3; }
    else if (f < 75) { int q = f - 48; ch = q / 9; int r = q % 9; y0 = (r / 3) * 4; x0 = (r % 3) * 4; ny = 4; }
    else if (f < 87) { int q = f - 75; ch = q / 4; int r = q % 4; y0 = (r / 2) * 6; x0 = (r % 2) * 6; ny = 6; }
    else             { ch = f - 87; y0 = 0; x0 = 0; ny = 12; }
    float m = -__builtin_inff();
    for (int y = 0; y < ny; ++y)
      for (int x = 0; x < ny; ++x)
        m = fmaxf(m, g8[ch * 144 + (y0 + y) * 12 + (x0 + x)]);
    dout[4096 + (size_t)img * 90 + f] = m;       // pooled output
    Apad[(size_t)img * 96 + f] = m;              // padded GEMM input
  } else if (tid < 96) {
    Apad[(size_t)img * 96 + tid] = 0.0f;         // K padding
  }
}

// ---------------- K2: fused relu(A@W1^T+b1) @ W7^T + b7, relu -> E (f32, bf16 WMMA) ---
__global__ __launch_bounds__(256) void emb_kernel(const float* __restrict__ Apad,
                                                  const __bf16* __restrict__ W1B,
                                                  const __bf16* __restrict__ W7B,
                                                  const float* __restrict__ b1,
                                                  const float* __restrict__ b7,
                                                  float* __restrict__ E) {
  __shared__ __bf16 As[16 * 104]; // 16x96 A tile (bf16), stride 104
  __shared__ __bf16 Cs[16 * 520]; // 16x512 relu'd GEMM1 chunk (bf16), stride 520
  const int tid = threadIdx.x, wave = tid >> 5, lane = tid & 31;
  const int col = lane & 15, khalf = lane >> 4, arow = lane & 15;
  const int m0 = blockIdx.x * 16;
  for (int i = tid; i < 16 * 96; i += 256) {
    int r = i / 96, k = i % 96;
    As[r * 104 + k] = (__bf16)Apad[(size_t)(m0 + r) * 96 + k];
  }
  __syncthreads();
  v8f eacc = {0, 0, 0, 0, 0, 0, 0, 0};
  const int n2 = (wave & 3) * 16 + col;
  for (int nc = 0; nc < 8; ++nc) {
    // GEMM1: 32 N-tiles over 8 waves, K = 96 (3 bf16 WMMAs per tile)
    for (int tt = 0; tt < 4; ++tt) {
      int tn = wave * 4 + tt;
      int n  = nc * 512 + tn * 16 + col;
      v8f acc = {0, 0, 0, 0, 0, 0, 0, 0};
      #pragma unroll
      for (int kb = 0; kb < 3; ++kb) {
        BF16x16 ua;
        int ab = arow * 104 + kb * 32 + 8 * khalf;
        ua.h[0] = *(const v8bf*)&As[ab];
        ua.h[1] = *(const v8bf*)&As[ab + 16];
        v16bf bv = *(const v16bf*)&W1B[(size_t)kb * 131072 + (size_t)n * 32 + khalf * 16];
        acc = wmma_bf(ua.v, bv, acc);
      }
      float bias = b1[n];
      #pragma unroll
      for (int v = 0; v < 8; ++v)
        Cs[(v + 8 * khalf) * 520 + tn * 16 + col] = (__bf16)fmaxf(acc[v] + bias, 0.0f);
    }
    __syncthreads();
    // GEMM2 accumulate: 4 N-tiles (waves 0-3), K = 512 chunk (16 WMMAs)
    if (wave < 4) {
      for (int kb2 = 0; kb2 < 16; ++kb2) {
        BF16x16 ua;
        int ab = arow * 520 + kb2 * 32 + 8 * khalf;
        ua.h[0] = *(const v8bf*)&Cs[ab];
        ua.h[1] = *(const v8bf*)&Cs[ab + 16];
        int kbG = nc * 16 + kb2;
        v16bf bv = *(const v16bf*)&W7B[(size_t)kbG * 2048 + n2 * 32 + khalf * 16];
        eacc = wmma_bf(ua.v, bv, eacc);
      }
    }
    __syncthreads();
  }
  if (wave < 4) {
    float bias = b7[n2];
    #pragma unroll
    for (int v = 0; v < 8; ++v) {
      int rw = v + 8 * khalf;
      E[(size_t)(m0 + rw) * 64 + n2] = fmaxf(eacc[v] + bias, 0.0f);
    }
  }
}

// ---------------- K3: X1 = E @ wih1^T + bih1 + bhh1 (exact f32 WMMA) -------
__global__ __launch_bounds__(256) void x1_kernel(const float* __restrict__ E,
                                                 const float* __restrict__ WIH1T,
                                                 const float* __restrict__ b1c,
                                                 float* __restrict__ X1) {
  __shared__ float Es[16 * 68];
  const int tid = threadIdx.x, wave = tid >> 5, lane = tid & 31;
  const int col = lane & 15, khalf = lane >> 4, arow = lane & 15;
  const int m0 = blockIdx.x * 16;
  for (int i = tid; i < 16 * 64; i += 256) {
    int r = i >> 6, k = i & 63;
    Es[r * 68 + k] = E[(size_t)(m0 + r) * 64 + k];
  }
  __syncthreads();
  for (int tt = 0; tt < 2; ++tt) {
    int n = (wave * 2 + tt) * 16 + col;
    v8f acc = {0, 0, 0, 0, 0, 0, 0, 0};
    for (int k0 = 0; k0 < 64; k0 += 4) {
      int kk = k0 + 2 * khalf;
      v2f a; a.x = Es[arow * 68 + kk]; a.y = Es[arow * 68 + kk + 1];
      v2f b; b.x = WIH1T[kk * 256 + n]; b.y = WIH1T[(kk + 1) * 256 + n];
      acc = wmma4(a, b, acc);
    }
    float bias = b1c[n];
    #pragma unroll
    for (int v = 0; v < 8; ++v) {
      int rw = v + 8 * khalf;
      X1[(size_t)(m0 + rw) * 256 + n] = acc[v] + bias;
    }
  }
}

// ---------------- K4: persistent 256-step LSTM scan (1 block, 32 waves) ----
__global__ __launch_bounds__(1024, 1) void scan_kernel(
    const float* __restrict__ X1, const __bf16* __restrict__ W1B, const __bf16* __restrict__ W7B,
    const float* __restrict__ WIH1T, const float* __restrict__ WHH1T,
    const float* __restrict__ WIH2T, const float* __restrict__ WHH2T,
    const float* __restrict__ FWT, const float* __restrict__ b1c, const float* __restrict__ b2c,
    const float* __restrict__ spp_b, const float* __restrict__ fc7_b,
    const float* __restrict__ fore_b, const float* __restrict__ fc8_w,
    const float* __restrict__ fc8_b, float* __restrict__ dout) {
  __shared__ __bf16 AfeB[16 * 104]; // fpool as bf16 (rows 8-15 & cols 90-95 stay zero)
  __shared__ __bf16 CsB[16 * 264];  // fe chunk 16x256 (bf16), stride 264
  __shared__ float fe2[16 * 68];    // fe after fc7 (16x64, f32)
  __shared__ float psum[4 * 16 * 16];
  __shared__ float h1[512], c1[512], h1n[512], c1n[512];
  __shared__ float h2[256], c2[256], h2n[256], c2n[256];
  __shared__ float fh1[512], fh2[256];
  __shared__ float g[2048], gb2[1024];
  const int tid = threadIdx.x, wave = tid >> 5, lane = tid & 31;
  const int col = lane & 15, khalf = lane >> 4, arow = lane & 15;

  for (int i = tid; i < 16 * 104; i += 1024) AfeB[i] = (__bf16)0.0f;
  for (int i = tid; i < 512;  i += 1024) { h1[i] = 0.0f; c1[i] = 0.0f; }
  for (int i = tid; i < 256;  i += 1024) { h2[i] = 0.0f; c2[i] = 0.0f; }
  __syncthreads();

  for (int s = 0; s < 256; ++s) {
    // A: g1 = X1[:,s] + h1 @ whh1^T
    for (int idx = tid; idx < 2048; idx += 1024) {
      int b = idx >> 8, j = idx & 255;
      float sum = X1[(size_t)(b * 256 + s) * 256 + j];
      for (int k = 0; k < 64; ++k) sum += h1[b * 64 + k] * WHH1T[k * 256 + j];
      g[idx] = sum;
    }
    __syncthreads();
    // B: lstm1 pointwise
    if (tid < 512) {
      int b = tid >> 6, j = tid & 63, o0 = b * 256 + j;
      float cc = sigm(g[o0 + 64]) * c1[tid] + sigm(g[o0]) * tanhf(g[o0 + 128]);
      c1n[tid] = cc; h1n[tid] = sigm(g[o0 + 192]) * tanhf(cc);
    }
    __syncthreads();
    // C: g2 = h1n @ wih2^T + h2 @ whh2^T + b
    {
      int b = tid >> 7, j = tid & 127;
      float sum = b2c[j];
      for (int k = 0; k < 64; ++k) sum += h1n[b * 64 + k] * WIH2T[k * 128 + j];
      for (int k = 0; k < 32; ++k) sum += h2[b * 32 + k] * WHH2T[k * 128 + j];
      gb2[tid] = sum;
    }
    __syncthreads();
    // D: lstm2 pointwise
    if (tid < 256) {
      int b = tid >> 5, j = tid & 31, o0 = b * 128 + j;
      float cc = sigm(gb2[o0 + 32]) * c2[tid] + sigm(gb2[o0]) * tanhf(gb2[o0 + 64]);
      c2n[tid] = cc; h2n[tid] = sigm(gb2[o0 + 96]) * tanhf(cc);
    }
    __syncthreads();
    // E: fpool = h2n @ fw^T + b (-> fpooled out f32, AfeB bf16) ; progress out
    if (tid < 720) {
      int b = tid / 90, j = tid % 90;
      float sum = fore_b[j];
      for (int k = 0; k < 32; ++k) sum += h2n[b * 32 + k] * FWT[k * 90 + j];
      dout[188416 + (size_t)(b * 256 + s) * 90 + j] = sum;
      AfeB[b * 104 + j] = (__bf16)sum;
    } else if (tid < 728) {
      int b = tid - 720;
      float d = fc8_b[0];
      for (int k = 0; k < 32; ++k) d += h2n[b * 32 + k] * fc8_w[k];
      dout[b * 256 + s] = sigm(d);
    }
    __syncthreads();
    // F: fe = relu(fpool@W1^T+b1); fe2 = relu(fe@W7^T+b7)  [bf16 WMMA chain]
    v8f facc = {0, 0, 0, 0, 0, 0, 0, 0};
    for (int nc = 0; nc < 16; ++nc) {
      if (wave < 16) { // GEMM1: 16 tiles x K=96 (3 WMMAs each)
        int n = nc * 256 + wave * 16 + col;
        v8f acc = {0, 0, 0, 0, 0, 0, 0, 0};
        #pragma unroll
        for (int kb = 0; kb < 3; ++kb) {
          BF16x16 ua;
          int ab = arow * 104 + kb * 32 + 8 * khalf;
          ua.h[0] = *(const v8bf*)&AfeB[ab];
          ua.h[1] = *(const v8bf*)&AfeB[ab + 16];
          v16bf bv = *(const v16bf*)&W1B[(size_t)kb * 131072 + (size_t)n * 32 + khalf * 16];
          acc = wmma_bf(ua.v, bv, acc);
        }
        float bias = spp_b[n];
        #pragma unroll
        for (int v = 0; v < 8; ++v)
          CsB[(v + 8 * khalf) * 264 + wave * 16 + col] = (__bf16)fmaxf(acc[v] + bias, 0.0f);
      }
      __syncthreads();
      if (wave < 8) { // GEMM2: 4 tiles, chunk-K split over 2 waves each
        int tle = wave & 3, kh = wave >> 2, n2 = tle * 16 + col;
        #pragma unroll
        for (int q = 0; q < 4; ++q) {
          int kb2 = kh * 4 + q;
          BF16x16 ua;
          int ab = arow * 264 + kb2 * 32 + 8 * khalf;
          ua.h[0] = *(const v8bf*)&CsB[ab];
          ua.h[1] = *(const v8bf*)&CsB[ab + 16];
          int kbG = nc * 8 + kb2;
          v16bf bv = *(const v16bf*)&W7B[(size_t)kbG * 2048 + n2 * 32 + khalf * 16];
          facc = wmma_bf(ua.v, bv, facc);
        }
      }
      __syncthreads();
    }
    if (wave >= 4 && wave < 8) {
      int tle = wave & 3;
      #pragma unroll
      for (int v = 0; v < 8; ++v) psum[(tle * 16 + v + 8 * khalf) * 16 + col] = facc[v];
    }
    __syncthreads();
    if (wave < 4) {
      #pragma unroll
      for (int v = 0; v < 8; ++v) {
        int rw = v + 8 * khalf;
        float x = facc[v] + psum[(wave * 16 + rw) * 16 + col] + fc7_b[wave * 16 + col];
        fe2[rw * 68 + wave * 16 + col] = fmaxf(x, 0.0f);
      }
    }
    __syncthreads();
    // G: fg1 = fe2 @ wih1^T + h1n @ whh1^T + b
    for (int idx = tid; idx < 2048; idx += 1024) {
      int b = idx >> 8, j = idx & 255;
      float sum = b1c[j];
      for (int k = 0; k < 64; ++k)
        sum += fe2[b * 68 + k] * WIH1T[k * 256 + j] + h1n[b * 64 + k] * WHH1T[k * 256 + j];
      g[idx] = sum;
    }
    __syncthreads();
    // H: forecast lstm1 pointwise (uses c1n, does not update state)
    if (tid < 512) {
      int b = tid >> 6, j = tid & 63, o0 = b * 256 + j;
      float cc = sigm(g[o0 + 64]) * c1n[tid] + sigm(g[o0]) * tanhf(g[o0 + 128]);
      fh1[tid] = sigm(g[o0 + 192]) * tanhf(cc);
    }
    __syncthreads();
    // I: fg2
    {
      int b = tid >> 7, j = tid & 127;
      float sum = b2c[j];
      for (int k = 0; k < 64; ++k) sum += fh1[b * 64 + k] * WIH2T[k * 128 + j];
      for (int k = 0; k < 32; ++k) sum += h2n[b * 32 + k] * WHH2T[k * 128 + j];
      gb2[tid] = sum;
    }
    __syncthreads();
    // J: forecast lstm2 pointwise
    if (tid < 256) {
      int b = tid >> 5, j = tid & 31, o0 = b * 128 + j;
      float cc = sigm(gb2[o0 + 32]) * c2n[tid] + sigm(gb2[o0]) * tanhf(gb2[o0 + 64]);
      fh2[tid] = sigm(gb2[o0 + 96]) * tanhf(cc);
    }
    __syncthreads();
    // K: fprogress out + state update
    if (tid < 8) {
      float d = fc8_b[0];
      for (int k = 0; k < 32; ++k) d += fh2[tid * 32 + k] * fc8_w[k];
      dout[2048 + tid * 256 + s] = sigm(d);
    }
    if (tid < 512) { h1[tid] = h1n[tid]; c1[tid] = c1n[tid]; }
    if (tid < 256) { h2[tid] = h2n[tid]; c2[tid] = c2n[tid]; }
    __syncthreads();
  }
}

// ---------------- launch ----------------
extern "C" void kernel_launch(void* const* d_in, const int* in_sizes, int n_in,
                              void* d_out, int out_size, void* d_ws, size_t ws_size,
                              hipStream_t stream) {
  (void)in_sizes; (void)n_in; (void)out_size; (void)ws_size;
  const float* frames   = (const float*)d_in[0];
  const float* spp_fc_w = (const float*)d_in[1];
  const float* spp_fc_b = (const float*)d_in[2];
  const float* fc7_w    = (const float*)d_in[3];
  const float* fc7_b    = (const float*)d_in[4];
  const float* l1_wih   = (const float*)d_in[5];
  const float* l1_whh   = (const float*)d_in[6];
  const float* l1_bih   = (const float*)d_in[7];
  const float* l1_bhh   = (const float*)d_in[8];
  const float* l2_wih   = (const float*)d_in[9];
  const float* l2_whh   = (const float*)d_in[10];
  const float* l2_bih   = (const float*)d_in[11];
  const float* l2_bhh   = (const float*)d_in[12];
  const float* fore_w   = (const float*)d_in[13];
  const float* fore_b   = (const float*)d_in[14];
  const float* fc8_w    = (const float*)d_in[15];
  const float* fc8_b    = (const float*)d_in[16];
  float* ws  = (float*)d_ws;
  float* out = (float*)d_out;
  const __bf16* W1B = (const __bf16*)(ws + WS_W1B);
  const __bf16* W7B = (const __bf16*)(ws + WS_W7B);

  prep_kernel<<<2749, 256, 0, stream>>>(spp_fc_w, fc7_w, l1_wih, l1_whh, l2_wih, l2_whh,
                                        fore_w, l1_bih, l1_bhh, l2_bih, l2_bhh, ws);
  spp_kernel<<<2048, 256, 0, stream>>>(frames, out, ws + WS_APAD);
  emb_kernel<<<128, 256, 0, stream>>>(ws + WS_APAD, W1B, W7B, spp_fc_b, fc7_b, ws + WS_E);
  x1_kernel<<<128, 256, 0, stream>>>(ws + WS_E, ws + WS_WIH1T, ws + WS_B1C, ws + WS_X1);
  scan_kernel<<<1, 1024, 0, stream>>>(ws + WS_X1, W1B, W7B,
                                      ws + WS_WIH1T, ws + WS_WHH1T, ws + WS_WIH2T, ws + WS_WHH2T,
                                      ws + WS_FWT, ws + WS_B1C, ws + WS_B2C,
                                      spp_fc_b, fc7_b, fore_b, fc8_w, fc8_b, out);
}
